// VectorQuantizer_45165876084990
// MI455X (gfx1250) — matrix operand order repaired and checked by reading, not compiled, not run on hardware
//
#include <hip/hip_runtime.h>

typedef __attribute__((ext_vector_type(16))) __bf16 v16bf;
typedef __attribute__((ext_vector_type(8)))  float  v8f;

#define DDIM  1024   // embedding dim
#define NPIX  256    // H*W

// ---------------------------------------------------------------------------
// Kernel 1: x [B, D, N] -> A_bf16 [token][D]  (transpose + f32->bf16)
// ---------------------------------------------------------------------------
__global__ __launch_bounds__(256) void k_transpose_bf16(
    const float* __restrict__ x, unsigned short* __restrict__ Abf) {
  __shared__ float tile[32][33];
  const int t  = threadIdx.x;
  const int tx = t & 31;
  const int ty = t >> 5;
  const int tok0 = blockIdx.x * 32;
  const int kd0  = blockIdx.y * 32;
  const int b  = tok0 >> 8;
  const int n0 = tok0 & 255;
  const float* xp = x + (size_t)b * DDIM * NPIX;
#pragma unroll
  for (int j = 0; j < 4; ++j) {
    int kd = kd0 + ty + j * 8;
    tile[ty + j * 8][tx] = xp[(size_t)kd * NPIX + n0 + tx];
  }
  __syncthreads();
#pragma unroll
  for (int j = 0; j < 4; ++j) {
    int tok = tok0 + ty + j * 8;
    __bf16 h = (__bf16)tile[tx][ty + j * 8];
    Abf[(size_t)tok * DDIM + kd0 + tx] = __builtin_bit_cast(unsigned short, h);
  }
}

// ---------------------------------------------------------------------------
// Kernel 2: e_sq[k] = ||codebook_k||^2  AND codebook f32 -> bf16 copy.
// One wave per row; reads are 128B coalesced, bf16 writes 64B coalesced.
// ---------------------------------------------------------------------------
__global__ __launch_bounds__(256) void k_esq_cvt(
    const float* __restrict__ cbk, float* __restrict__ esq,
    unsigned short* __restrict__ Bbf) {
  const int w    = threadIdx.x >> 5;
  const int lane = threadIdx.x & 31;
  const int row  = blockIdx.x * 8 + w;
  const float* p = cbk + (size_t)row * DDIM;
  unsigned short* q = Bbf + (size_t)row * DDIM;
  float s = 0.f;
#pragma unroll 4
  for (int i = 0; i < DDIM / 32; ++i) {
    float v = p[lane + 32 * i];
    s += v * v;
    q[lane + 32 * i] = __builtin_bit_cast(unsigned short, (__bf16)v);
  }
#pragma unroll
  for (int off = 16; off; off >>= 1) s += __shfl_xor(s, off, 32);
  if (lane == 0) esq[row] = s;
}

// ---------------------------------------------------------------------------
// Kernel 3: bf16 WMMA GEMM (cross term) + running argmin per token.
// Grid: 32 M-tiles (128 tokens) x 8 N-chunks (1024 codes), 256 threads.
// Macro tile 128x256; wave w: (w&1)->64-row M block, (w>>1)->64-col N block;
// each wave computes a 64x64 f32 tile = 16 v_wmma per 32-deep k-step.
// Staging: double-buffered LDS filled with global_load_async_to_lds_b128
// (ASYNCcnt), overlapped with WMMA; one barrier per k-step.
// LDS rows padded to 80B -> conflict-free ds_load_b128 fragment reads.
// ---------------------------------------------------------------------------
__global__ __launch_bounds__(256) void k_vq(
    const unsigned short* __restrict__ Abf,
    const unsigned short* __restrict__ Bbf,
    const float* __restrict__ esq,
    unsigned long long* __restrict__ partial) {
  __shared__ __align__(16) unsigned short At[2][128 * 40];
  __shared__ __align__(16) unsigned short Bt[2][256 * 40];
  __shared__ float esh[256];
  __shared__ unsigned long long best[128];

  const int t      = threadIdx.x;
  const int mtile  = blockIdx.x;   // 0..31
  const int nchunk = blockIdx.y;   // 0..7
  const int lane   = t & 31;
  const int w      = t >> 5;
  const int wm     = w & 1;        // M block (64 rows)
  const int wn     = w >> 1;       // N block (64 cols)
  const int halfl  = lane >> 4;
  const int lr     = lane & 15;
  const int srow   = t >> 2;       // staging: 64 rows per pass
  const int sch    = t & 3;        // staging: 16B chunk within 64B row

  if (t < 128) best[t] = ~0ull;

  for (int nn = 0; nn < 4; ++nn) {
    const int cbase = nchunk * 1024 + nn * 256;  // first code of macro tile
    __syncthreads();                             // prev nn done with esh/LDS
    esh[t] = esq[cbase + t];

    // ---- async staging helpers: raw bf16 copy, 16B per lane ----
    auto stageA = [&](int kk, int buf) {
#pragma unroll
      for (int i = 0; i < 2; ++i) {
        int row = i * 64 + srow;                 // 128 rows
        unsigned goff = (unsigned)(((mtile * 128 + row) * DDIM + kk) * 2 + sch * 16);
        unsigned loff = (unsigned)(unsigned long long)&At[buf][row * 40 + sch * 8];
        asm volatile("global_load_async_to_lds_b128 %0, %1, %2"
                     :: "v"(loff), "v"(goff), "s"(Abf) : "memory");
      }
    };
    auto stageB = [&](int kk, int buf) {
#pragma unroll
      for (int i = 0; i < 4; ++i) {
        int row = i * 64 + srow;                 // 256 rows
        unsigned goff = (unsigned)(((cbase + row) * DDIM + kk) * 2 + sch * 16);
        unsigned loff = (unsigned)(unsigned long long)&Bt[buf][row * 40 + sch * 8];
        asm volatile("global_load_async_to_lds_b128 %0, %1, %2"
                     :: "v"(loff), "v"(goff), "s"(Bbf) : "memory");
      }
    };

    stageA(0, 0);
    stageB(0, 0);
    v8f acc[4][4] = {};
    asm volatile("s_wait_asynccnt 0x0" ::: "memory");
    __syncthreads();

    for (int k = 0; k < DDIM; k += 32) {
      const int cur = (k >> 5) & 1;
      if (k + 32 < DDIM) {                       // prefetch next tile (async)
        stageA(k + 32, cur ^ 1);
        stageB(k + 32, cur ^ 1);
      }
      union V16 { uint4 q[2]; v16bf v; };
      V16 af[4], bfr[4];
#pragma unroll
      for (int mt = 0; mt < 4; ++mt) {
        const unsigned short* p =
            &At[cur][(wm * 64 + mt * 16 + lr) * 40 + halfl * 8];
        af[mt].q[0] = *(const uint4*)(p);        // K 0..7  (lo)/ 8..15 (hi)
        af[mt].q[1] = *(const uint4*)(p + 16);   // K 16..23(lo)/ 24..31(hi)
      }
#pragma unroll
      for (int nt = 0; nt < 4; ++nt) {
        const unsigned short* p =
            &Bt[cur][(wn * 64 + nt * 16 + lr) * 40 + halfl * 8];
        bfr[nt].q[0] = *(const uint4*)(p);       // K 0..7  (lo)/ 8..15 (hi)
        bfr[nt].q[1] = *(const uint4*)(p + 16);  // K 16..23(lo)/ 24..31(hi)
      }
#pragma unroll
      for (int mt = 0; mt < 4; ++mt)
#pragma unroll
        for (int nt = 0; nt < 4; ++nt)
          acc[mt][nt] = __builtin_amdgcn_wmma_f32_16x16x32_bf16(
              false, af[mt].v, false, bfr[nt].v,
              (short)0, acc[mt][nt], false, false);
      asm volatile("s_wait_asynccnt 0x0" ::: "memory");
      __syncthreads();
    }

    // ---- epilogue: dist = e_sq - 2*cross, argmin via packed u64 LDS min ----
#pragma unroll
    for (int mt = 0; mt < 4; ++mt) {
#pragma unroll
      for (int r = 0; r < 8; ++r) {
        int mloc = wm * 64 + mt * 16 + halfl * 8 + r;
        unsigned long long kb = ~0ull;
#pragma unroll
        for (int nt = 0; nt < 4; ++nt) {
          int nloc = wn * 64 + nt * 16 + lr;
          float dist = esh[nloc] - 2.0f * acc[mt][nt][r];
          unsigned int u = __float_as_uint(dist);
          u = (u & 0x80000000u) ? ~u : (u | 0x80000000u);  // order-preserving
          unsigned long long key =
              ((unsigned long long)u << 32) | (unsigned)(cbase + nloc);
          kb = key < kb ? key : kb;
        }
        atomicMin(&best[mloc], kb);              // ds_min_u64
      }
    }
  }
  __syncthreads();
  if (t < 128)
    partial[(size_t)(mtile * 128 + t) * 8 + nchunk] = best[t];
}

// ---------------------------------------------------------------------------
// Kernel 4: reduce 8 partials per token, gather values[idx], write [B,D,H,W]
// through a 16x16 LDS transpose so both global sides are coalesced.
// ---------------------------------------------------------------------------
__global__ __launch_bounds__(256) void k_gather(
    const unsigned long long* __restrict__ partial,
    const float* __restrict__ values,
    float* __restrict__ out) {
  __shared__ unsigned long long sl[16];
  __shared__ int sidx[16];
  __shared__ float tile[16][17];
  const int t  = threadIdx.x;
  const int t0 = blockIdx.x * 16;
  const int b  = t0 >> 8;
  const int n0 = t0 & 255;
  if (t < 16) sl[t] = ~0ull;
  __syncthreads();
  if (t < 128)
    atomicMin(&sl[t >> 3], partial[(size_t)(t0 + (t >> 3)) * 8 + (t & 7)]);
  __syncthreads();
  if (t < 16) sidx[t] = (int)(unsigned)(sl[t] & 0xFFFFFFFFull);
  __syncthreads();
  const int nl = t >> 4, dl = t & 15;
  const int dd = t >> 4, nw = t & 15;
  float* ob = out + (size_t)b * DDIM * NPIX + n0;
  for (int dblk = 0; dblk < 64; ++dblk) {
    tile[nl][dl] = values[(size_t)sidx[nl] * DDIM + dblk * 16 + dl];
    __syncthreads();
    ob[(size_t)(dblk * 16 + dd) * NPIX + nw] = tile[nw][dd];
    __syncthreads();
  }
}

// ---------------------------------------------------------------------------
extern "C" void kernel_launch(void* const* d_in, const int* in_sizes, int n_in,
                              void* d_out, int out_size, void* d_ws,
                              size_t ws_size, hipStream_t stream) {
  (void)in_sizes; (void)n_in; (void)out_size; (void)ws_size;
  const float* x    = (const float*)d_in[0];   // [16,1024,16,16]
  const float* cbk  = (const float*)d_in[1];   // [1,8192,1024]
  const float* vals = (const float*)d_in[2];   // [1,8192,1024]
  float* out = (float*)d_out;                  // [16,1024,16,16]

  char* ws = (char*)d_ws;
  unsigned short*     Abf     = (unsigned short*)(ws);              //  8 MB
  unsigned short*     Bbf     = (unsigned short*)(ws + 8388608);    // 16 MB
  unsigned long long* partial = (unsigned long long*)(ws + 25165824); // 256 KB
  float*              esq     = (float*)(ws + 25427968);            // 32 KB

  k_transpose_bf16<<<dim3(128, 32), 256, 0, stream>>>(x, Abf);
  k_esq_cvt<<<1024, 256, 0, stream>>>(cbk, esq, Bbf);
  k_vq<<<dim3(32, 8), 256, 0, stream>>>(Abf, Bbf, esq, partial);
  k_gather<<<256, 256, 0, stream>>>(partial, vals, out);
}
